// CRFModel_19439021982259
// MI455X (gfx1250) — matrix operand order repaired and checked by reading, not compiled, not run on hardware
//
#include <hip/hip_runtime.h>
#include <hip/hip_bf16.h>

#define BB 64
#define TT 512
#define DD 1024
#define UU 128

typedef __attribute__((ext_vector_type(16))) __bf16 v16bf;
typedef __attribute__((ext_vector_type(8)))  float  v8f;

// ---------------------------------------------------------------------------
// CDNA5 async copy: 16B global -> LDS, tracked by ASYNCcnt (ISA §10, §15.18).
// VDST = per-lane LDS byte address, VADDR = 64-bit global address ("GV" mode).
// ---------------------------------------------------------------------------
__device__ __forceinline__ void async_copy16(const __bf16* g, const __bf16* l) {
    unsigned lds_addr = (unsigned)(size_t)l;          // low 32 bits = LDS offset
    unsigned long long ga = (unsigned long long)(size_t)g;
    asm volatile("global_load_async_to_lds_b128 %0, %1, off"
                 :: "v"(lds_addr), "v"(ga) : "memory");
}

// ---------------------------------------------------------------------------
// Kernel 0: one-time W prep: Wt[n][k] = (bf16)W[k][n]  (256 KB, L2-resident).
// Layout matches the WMMA B-fragment order so GEMM staging is a pure copy.
// ---------------------------------------------------------------------------
__global__ __launch_bounds__(256) void wprep(const float* __restrict__ dk,
                                             __bf16* __restrict__ wt)
{
    const int n = blockIdx.x;                          // 128 blocks
    for (int k = threadIdx.x; k < DD; k += 256)
        wt[(size_t)n * DD + k] = (__bf16)dk[(size_t)k * UU + n];
}

// ---------------------------------------------------------------------------
// Kernel 1: p = x @ W + bias (+ boundaries) via v_wmma_f32_16x16x32_bf16.
// 256 threads (8 waves), 128x128 tile. A: direct HBM loads (no reuse).
// B: async-LDS double-buffered 8KB tiles from pre-converted Wt.
// ---------------------------------------------------------------------------
__global__ __launch_bounds__(256) void potentials_gemm(
    const float* __restrict__ x, const __bf16* __restrict__ wt,
    const float* __restrict__ bias, const float* __restrict__ lb,
    const float* __restrict__ rb, const int* __restrict__ lengths,
    float* __restrict__ p)
{
    __shared__ __align__(32) __bf16 lB[2][UU * 32];    // [buf][n][k] 2 x 8KB

    const int tid  = threadIdx.x;
    const int wave = tid >> 5;
    const int lane = tid & 31;
    const int half = lane >> 4;
    const int l16  = lane & 15;
    const int rowBase = blockIdx.x * 128;

    // per-thread async chunk coords: 512 x 16B chunks / stage, 2 per thread
    const int c0 = tid,      n0 = c0 >> 2, o0 = (c0 & 3) * 8;  // element offs
    const int c1 = tid + 256, n1 = c1 >> 2, o1 = (c1 & 3) * 8;

    // this wave's x row (lanes l and l+16 share a row; halves split K)
    const float* arow = x + (size_t)(rowBase + wave * 16 + l16) * DD;

    v8f acc[8];
#pragma unroll
    for (int j = 0; j < 8; ++j) acc[j] = (v8f)(0.0f);

    // prologue: stage k0 = 0 into buffer 0
    async_copy16(wt + (size_t)n0 * DD + o0, &lB[0][n0 * 32 + o0]);
    async_copy16(wt + (size_t)n1 * DD + o1, &lB[0][n1 * 32 + o1]);

    for (int i = 0; i < DD / 32; ++i) {
        const int k0 = i * 32;

        // issue next B tile while we compute on the current one
        if (i + 1 < DD / 32) {
            const int kn = k0 + 32;
            async_copy16(wt + (size_t)n0 * DD + kn + o0,
                         &lB[(i + 1) & 1][n0 * 32 + o0]);
            async_copy16(wt + (size_t)n1 * DD + kn + o1,
                         &lB[(i + 1) & 1][n1 * 32 + o1]);
        }

        // A fragment (16x32 bf16): four 16B f32 loads per lane straight from HBM,
        // contiguous layout: K = h*16..h*16+15 (matches B)
        v16bf a;
        {
            const float4* ap  = (const float4*)(arow + k0 + half * 16);
            float fa[16];
            *(float4*)&fa[0]  = ap[0];
            *(float4*)&fa[4]  = ap[1];
            *(float4*)&fa[8]  = ap[2];
            *(float4*)&fa[12] = ap[3];
#pragma unroll
            for (int q = 0; q < 16; ++q) a[q] = (__bf16)fa[q];
        }

        // async loads complete in order: <=2 outstanding => stage i has landed
        if (i + 1 < DD / 32) asm volatile("s_wait_asynccnt 0x2" ::: "memory");
        else                 asm volatile("s_wait_asynccnt 0x0" ::: "memory");
        __syncthreads();

        const __bf16* bbuf = lB[i & 1];
#pragma unroll
        for (int j = 0; j < 8; ++j) {
            // B (32x16): lane = column n, half selects K 0-15 / 16-31
            v16bf bfrag = *(const v16bf*)&bbuf[(j * 16 + l16) * 32 + half * 16];
            acc[j] = __builtin_amdgcn_wmma_f32_16x16x32_bf16(
                false, a, false, bfrag, (short)0, acc[j], false, false);
        }
        __syncthreads();   // protect buffer (i&1) before stage i+2 overwrites
    }

    // ---- epilogue: bias + boundary terms, store f32 ----
    const int bb   = rowBase / TT;        // 128 | 512 -> whole block same batch
    const int lenb = lengths[bb];
#pragma unroll
    for (int j = 0; j < 8; ++j) {
        const int n = j * 16 + l16;
        const float bn = bias[n], lbn = lb[n], rbn = rb[n];
#pragma unroll
        for (int v = 0; v < 8; ++v) {
            const int m = rowBase + wave * 16 + half * 8 + v;  // C layout
            const int t = m & (TT - 1);
            float val = acc[j][v] + bn;
            if (t == 0)        val += lbn;
            if (t == lenb - 1) val += rbn;
            p[(size_t)m * UU + n] = val;
        }
    }
}

// ---------------------------------------------------------------------------
// Kernel 2: per-batch CRF scan. 1 block / batch, 1 lane / state (128 thr).
// chain^T (bank-padded), alphas and all backpointers live in LDS (~134 KB of
// the 320 KB/WGP budget) so Viterbi backtrack never touches global memory.
// ---------------------------------------------------------------------------
__global__ __launch_bounds__(128) void crf_chain(
    const float* __restrict__ p, const float* __restrict__ chain,
    const int* __restrict__ lengths, const int* __restrict__ tags,
    float* __restrict__ dec, float* __restrict__ lenout,
    float* __restrict__ nll)
{
    __shared__ float chainT[UU * 130];          // chainT[u*130+j] = chain[j][u]
    __shared__ float sAv[UU];                   // viterbi alpha
    __shared__ float sAl[UU];                   // logsumexp alpha
    __shared__ float red[UU];
    __shared__ int   redi[UU];
    __shared__ unsigned char bpL[(TT - 1) * UU];

    const int u   = threadIdx.x;
    const int b   = blockIdx.x;
    const int len = lengths[b];

    for (int c = 0; c < UU; ++c)                // thread u reads chain row u
        chainT[c * 130 + u] = chain[u * UU + c];

    const float* pb = p + (size_t)b * TT * UU;
    sAv[u] = pb[u];
    sAl[u] = pb[u];
    __syncthreads();

    for (int t = 1; t < TT; ++t) {
        float newv = 0.f, newl = 0.f; int bi = 0;
        const bool act = (t < len);
        if (act) {
            const float* cu = &chainT[u * 130];
            float bestv = -3.402823466e38f;
            float ml = -3.402823466e38f, sl = 0.f;
#pragma unroll 8
            for (int j = 0; j < UU; ++j) {
                const float c  = cu[j];
                const float sv = sAv[j] + c;
                if (sv > bestv) { bestv = sv; bi = j; }     // first-max tiebreak
                const float se = sAl[j] + c;
                if (se > ml) { sl = sl * __expf(ml - se) + 1.f; ml = se; }
                else           sl += __expf(se - ml);
            }
            const float pot = pb[t * UU + u];
            newv = bestv + pot;
            newl = ml + __logf(sl) + pot;
        }
        __syncthreads();
        if (act) {
            sAv[u] = newv;
            sAl[u] = newl;
            bpL[(t - 1) * UU + u] = (unsigned char)bi;
        }
        __syncthreads();
    }

    // ---- log_norm = logsumexp_u(sAl) ----
    red[u] = sAl[u];
    __syncthreads();
    for (int s = 64; s > 0; s >>= 1) {
        if (u < s) red[u] = fmaxf(red[u], red[u + s]);
        __syncthreads();
    }
    const float mx = red[0];
    __syncthreads();
    red[u] = __expf(sAl[u] - mx);
    __syncthreads();
    for (int s = 64; s > 0; s >>= 1) {
        if (u < s) red[u] += red[u + s];
        __syncthreads();
    }
    const float log_norm = mx + __logf(red[0]);
    __syncthreads();

    // ---- last = argmax_u(sAv), first index on ties ----
    red[u] = sAv[u]; redi[u] = u;
    __syncthreads();
    for (int s = 64; s > 0; s >>= 1) {
        if (u < s) {
            const float v2 = red[u + s]; const int i2 = redi[u + s];
            if (v2 > red[u] || (v2 == red[u] && i2 < redi[u])) {
                red[u] = v2; redi[u] = i2;
            }
        }
        __syncthreads();
    }
    const int last = redi[0];
    __syncthreads();

    // ---- gold sequence score (unary + transitions) ----
    const int* tg = tags + b * TT;
    float sc = 0.f;
    for (int t = u; t < TT; t += UU)
        if (t < len) sc += pb[t * UU + tg[t]];
    for (int t = u + 1; t < TT; t += UU)
        if (t < len) sc += chainT[tg[t] * 130 + tg[t - 1]];   // chain[prev][cur]
    red[u] = sc;
    __syncthreads();
    for (int s = 64; s > 0; s >>= 1) {
        if (u < s) red[u] += red[u + s];
        __syncthreads();
    }

    if (u == 0) {
        nll[b]    = log_norm - red[0];
        lenout[b] = (float)len;
        // Viterbi backtrack entirely inside LDS
        int cur = last;
        float* db = dec + b * TT;
        for (int t = TT - 1; t >= 1; --t) {
            if (t < len) { db[t] = (float)cur; cur = bpL[(t - 1) * UU + cur]; }
            else         { db[t] = 0.f; }
        }
        db[0] = (float)cur;   // len >= 1 always
    }
}

// ---------------------------------------------------------------------------
extern "C" void kernel_launch(void* const* d_in, const int* in_sizes, int n_in,
                              void* d_out, int out_size, void* d_ws, size_t ws_size,
                              hipStream_t stream) {
    (void)in_sizes; (void)n_in; (void)out_size; (void)ws_size;

    const float* x       = (const float*)d_in[0];
    const int*   lengths = (const int*)  d_in[1];
    const int*   tags    = (const int*)  d_in[2];
    const float* dk      = (const float*)d_in[3];
    const float* bias    = (const float*)d_in[4];
    const float* chain   = (const float*)d_in[5];
    const float* lb      = (const float*)d_in[6];
    const float* rb      = (const float*)d_in[7];

    float* out    = (float*)d_out;
    float* dec    = out;                               // B*T       decoded
    float* p      = out + (size_t)BB * TT;             // B*T*U     potentials
    float* lenout = p + (size_t)BB * TT * UU;          // B         lengths
    float* nllout = lenout + BB;                       // B         nll

    __bf16* wt = (__bf16*)d_ws;                        // 256 KB bf16 W^T

    wprep<<<UU, 256, 0, stream>>>(dk, wt);
    potentials_gemm<<<(BB * TT) / 128, 256, 0, stream>>>(x, wt, bias, lb, rb,
                                                         lengths, p);
    crf_chain<<<BB, 128, 0, stream>>>(p, chain, lengths, tags,
                                      dec, lenout, nllout);
}
